// MultiHeadMuonLoRALinear_44014824849821
// MI455X (gfx1250) — matrix operand order repaired and checked by reading, not compiled, not run on hardware
//
#include <hip/hip_runtime.h>

// ---------------------------------------------------------------------------
// MI455X / gfx1250: MultiHead Muon-LoRA linear.
// All heavy math routed through v_wmma_f32_16x16x32_bf16 (wave32 WMMA).
// v2: 64x64 wave tiles (block 128x256) -> 16 WMMAs per wave per K-step vs
//     16 ds_load_b128, doubling FLOPs per LDS byte over v1.
// ---------------------------------------------------------------------------

typedef unsigned short ushort_t;
typedef __attribute__((ext_vector_type(16))) __bf16 v16bf;
typedef __attribute__((ext_vector_type(8)))  float  v8f;

__device__ __forceinline__ ushort_t f2bf(float f) {
    unsigned u = __float_as_uint(f);
    u += 0x7FFFu + ((u >> 16) & 1u);          // round-to-nearest-even
    return (ushort_t)(u >> 16);
}
__device__ __forceinline__ float bf2f(ushort_t b) {
    return __uint_as_float(((unsigned)b) << 16);
}

#define TILE_M 128
#define TILE_N 256
#define TILE_K 32
#define LDSW   40   // LDS row stride in ushorts (80 B => 16B-aligned, spreads banks)

// D[M,N] = s1 * (A @ opB) + s2 * Cin (+ bias).  A:[M,K] bf16 row-major.
// TRANSB: B is [N,K] row-major (compute A @ B^T) -> contiguous staging.
// else  : B is [K,N] row-major.
// OUTF32: write float, else bf16.
template <bool TRANSB, bool OUTF32>
__global__ __launch_bounds__(256)
void gemm_bf16(const ushort_t* __restrict__ A,
               const ushort_t* __restrict__ B,
               const ushort_t* __restrict__ Cin,   // nullable
               const float*    __restrict__ bias,  // nullable
               void* __restrict__ Dv,
               int M, int N, int K, float s1, float s2)
{
    __shared__ __align__(16) ushort_t lsA[TILE_M * LDSW];
    __shared__ __align__(16) ushort_t lsB[TILE_N * LDSW];

    const int tid   = threadIdx.x;
    const int lane  = tid & 31;
    const int wave  = tid >> 5;
    const int waveM = wave >> 2;              // 0..1  (64 rows each)
    const int waveN = wave & 3;               // 0..3  (64 cols each)
    const int bM    = blockIdx.y * TILE_M;
    const int bN    = blockIdx.x * TILE_N;

    const int rsel  = lane & 15;
    const int khalf = lane >> 4;

    v8f acc[4][4];
#pragma unroll
    for (int i = 0; i < 4; ++i)
#pragma unroll
        for (int j = 0; j < 4; ++j)
#pragma unroll
            for (int v = 0; v < 8; ++v) acc[i][j][v] = 0.0f;

    // staging decomposition
    const int rowSeg = tid >> 1;              // 0..127  (A tile: 16 elems/thread)
    const int kSeg   = (tid & 1) * 16;        // 0 or 16
    const int kRow   = tid >> 3;              // 0..31   (normal-B path)
    const int nSeg   = (tid & 7) * 32;        // 0..224  (normal-B: 32 elems/thread)

    union Fr { v16bf v; uint4 u4[2]; };

    for (int kt = 0; kt < K; kt += TILE_K) {
        // ---- stage A tile: lsA[row][k], contiguous both sides ----
        {
            const uint4* src = (const uint4*)(A + (size_t)(bM + rowSeg) * K + kt + kSeg);
            uint4 a0 = src[0], a1 = src[1];
            *(uint4*)(&lsA[rowSeg * LDSW + kSeg])     = a0;
            *(uint4*)(&lsA[rowSeg * LDSW + kSeg + 8]) = a1;
            if (kt + TILE_K < K)   // emits global_prefetch_b8 for next K tile
                __builtin_prefetch(A + (size_t)(bM + rowSeg) * K + kt + TILE_K + kSeg, 0, 1);
        }
        // ---- stage B tile as lsB[n][k] (n-major => frag reads are b128) ----
        if (TRANSB) {
            // 256 rows x 32 k: 32 elems (64 B) per thread, fully contiguous
            const uint4* src = (const uint4*)(B + (size_t)(bN + tid) * K + kt);
            uint4 b0 = src[0], b1 = src[1], b2 = src[2], b3 = src[3];
            uint4* dst = (uint4*)(&lsB[tid * LDSW]);
            dst[0] = b0; dst[1] = b1; dst[2] = b2; dst[3] = b3;
        } else {
            // 32 k-rows x 256 n: 32 contiguous n per thread, scatter into LDS
            const uint4* src = (const uint4*)(B + (size_t)(kt + kRow) * N + bN + nSeg);
            union { uint4 u4[4]; ushort_t s[32]; } tmp;
            tmp.u4[0] = src[0]; tmp.u4[1] = src[1];
            tmp.u4[2] = src[2]; tmp.u4[3] = src[3];
#pragma unroll
            for (int e = 0; e < 32; ++e)
                lsB[(nSeg + e) * LDSW + kRow] = tmp.s[e];
        }
        __syncthreads();

        // ---- fragments (ISA 16-bit A/B layouts) + 16 WMMAs per wave ----
        Fr bf[4];
#pragma unroll
        for (int j = 0; j < 4; ++j) {
            const ushort_t* p = &lsB[(waveN * 64 + j * 16 + rsel) * LDSW];
            bf[j].u4[0] = *(const uint4*)(p + khalf * 8);
            bf[j].u4[1] = *(const uint4*)(p + 16 + khalf * 8);
        }
#pragma unroll
        for (int i = 0; i < 4; ++i) {
            Fr af;
            const ushort_t* p = &lsA[(waveM * 64 + i * 16 + rsel) * LDSW];
            af.u4[0] = *(const uint4*)(p + khalf * 8);        // K 0-7 / 8-15
            af.u4[1] = *(const uint4*)(p + 16 + khalf * 8);   // K 16-23 / 24-31
#pragma unroll
            for (int j = 0; j < 4; ++j)
                acc[i][j] = __builtin_amdgcn_wmma_f32_16x16x32_bf16(
                    false, af.v, false, bf[j].v, (short)0, acc[i][j], false, false);
        }
        __syncthreads();
    }

    // ---- epilogue: C/D layout: VGPR v -> M = v + 8*khalf, lane -> N ----
    const bool hasC = (Cin != nullptr);
    const bool hasBias = (bias != nullptr);
#pragma unroll
    for (int i = 0; i < 4; ++i) {
        const int gr0 = bM + waveM * 64 + i * 16 + khalf * 8;
#pragma unroll
        for (int j = 0; j < 4; ++j) {
            const int gc = bN + waveN * 64 + j * 16 + rsel;
            const float bv = hasBias ? bias[gc] : 0.0f;
#pragma unroll
            for (int v = 0; v < 8; ++v) {
                const size_t o = (size_t)(gr0 + v) * N + gc;
                float val = s1 * acc[i][j][v] + bv;
                if (hasC) val += s2 * bf2f(Cin[o]);
                if (OUTF32) ((float*)Dv)[o] = val;
                else        ((ushort_t*)Dv)[o] = f2bf(val);
            }
        }
    }
}

// ---------------------------------------------------------------------------
// Small helper kernels
// ---------------------------------------------------------------------------

__global__ void zero_sums(float* s) {
    if (threadIdx.x < 12) s[threadIdx.x] = 0.0f;
}

// G = lora_B[:, h*4:h*4+4] @ lora_A[h*4:h*4+4, :]  (rank-4), bf16 out,
// plus Frobenius sum-of-squares reduction into *sumsq.
__global__ __launch_bounds__(256)
void build_g(const float* __restrict__ lA, const float* __restrict__ lB,
             ushort_t* __restrict__ X, float* __restrict__ sumsq, int h)
{
    const size_t idx = (size_t)blockIdx.x * 256 + threadIdx.x;  // over 4096*1024
    const int o  = (int)(idx >> 10);
    const int i0 = ((int)idx & 1023) << 2;
    const float b0 = lB[o * 16 + h * 4 + 0];
    const float b1 = lB[o * 16 + h * 4 + 1];
    const float b2 = lB[o * 16 + h * 4 + 2];
    const float b3 = lB[o * 16 + h * 4 + 3];
    const float* a0 = lA + (size_t)(h * 4 + 0) * 4096;
    const float* a1 = lA + (size_t)(h * 4 + 1) * 4096;
    const float* a2 = lA + (size_t)(h * 4 + 2) * 4096;
    const float* a3 = lA + (size_t)(h * 4 + 3) * 4096;

    float ss = 0.0f;
#pragma unroll
    for (int j = 0; j < 4; ++j) {
        const int i = i0 + j;
        float g = b0 * a0[i] + b1 * a1[i] + b2 * a2[i] + b3 * a3[i];
        X[(size_t)o * 4096 + i] = f2bf(g);
        ss += g * g;
    }
    // wave32 reduction + cross-wave via LDS + one atomic per block
#pragma unroll
    for (int off = 16; off > 0; off >>= 1) ss += __shfl_down(ss, off, 32);
    __shared__ float red[8];
    if ((threadIdx.x & 31) == 0) red[threadIdx.x >> 5] = ss;
    __syncthreads();
    if (threadIdx.x == 0) {
        float t = 0.0f;
#pragma unroll
        for (int w = 0; w < 8; ++w) t += red[w];
        atomicAdd(sumsq, t);
    }
}

__global__ void finalize_norm(const float* ss, float* alpha, float* inv) {
    float a = sqrtf(*ss);
    *alpha = a;
    *inv = 1.0f / (a + 1e-7f);
}

__global__ __launch_bounds__(256)
void scale_x(ushort_t* __restrict__ X, const float* __restrict__ inv) {
    const float s = *inv;
    const size_t i = ((size_t)blockIdx.x * 256 + threadIdx.x) * 4;
#pragma unroll
    for (int j = 0; j < 4; ++j) X[i + j] = f2bf(bf2f(X[i + j]) * s);
}

__global__ __launch_bounds__(256)
void cvt_bf16(const float* __restrict__ src, ushort_t* __restrict__ dst, size_t n) {
    const size_t i = ((size_t)blockIdx.x * 256 + threadIdx.x) * 4;
    if (i + 3 < n) {
        float4 v = *(const float4*)(src + i);
        dst[i + 0] = f2bf(v.x); dst[i + 1] = f2bf(v.y);
        dst[i + 2] = f2bf(v.z); dst[i + 3] = f2bf(v.w);
    }
}

// W_eff += alpha_h * X_h   (bf16 accumulate)
__global__ __launch_bounds__(256)
void add_weff(ushort_t* __restrict__ W, const ushort_t* __restrict__ X,
              const float* __restrict__ alpha) {
    const float a = *alpha;
    const size_t i = ((size_t)blockIdx.x * 256 + threadIdx.x) * 4;
#pragma unroll
    for (int j = 0; j < 4; ++j)
        W[i + j] = f2bf(bf2f(W[i + j]) + a * bf2f(X[i + j]));
}

// ---------------------------------------------------------------------------
// Host orchestration (stream-ordered, graph-capturable: no malloc/sync)
// ---------------------------------------------------------------------------

extern "C" void kernel_launch(void* const* d_in, const int* in_sizes, int n_in,
                              void* d_out, int out_size, void* d_ws, size_t ws_size,
                              hipStream_t stream)
{
    (void)in_sizes; (void)n_in; (void)out_size; (void)ws_size;

    const float* x    = (const float*)d_in[0];   // [4,2048,4096] -> [8192,4096]
    const float* W    = (const float*)d_in[1];   // [4096,4096]
    const float* bias = (const float*)d_in[2];   // [4096]
    const float* lA   = (const float*)d_in[3];   // [16,4096]
    const float* lB   = (const float*)d_in[4];   // [4096,16]

    const int D  = 4096;
    const int BS = 8192;
    const size_t SZ = (size_t)D * D;

    char* p = (char*)d_ws;
    ushort_t* slotA = (ushort_t*)p; p += SZ * 2;            // X ping
    ushort_t* slotB = (ushort_t*)p; p += SZ * 2;            // X pong
    ushort_t* Am    = (ushort_t*)p; p += SZ * 2;            // A = X X^T
    ushort_t* Bm    = (ushort_t*)p; p += SZ * 2;            // B = b*A + c*A@A
    ushort_t* xb    = (ushort_t*)p; p += (size_t)BS * D * 2; // bf16(x)
    ushort_t* Weff  = (ushort_t*)p; p += SZ * 2;            // W + sum alpha*M
    float*    sums  = (float*)p;                            // [4] ss, [4] alpha, [4] inv

    zero_sums<<<1, 32, 0, stream>>>(sums);
    cvt_bf16<<<(int)(((size_t)BS * D / 4 + 255) / 256), 256, 0, stream>>>(x, xb, (size_t)BS * D);
    cvt_bf16<<<(int)((SZ / 4 + 255) / 256), 256, 0, stream>>>(W, Weff, SZ);

    const float nsA = 3.4445f, nsB = -4.775f, nsC = 2.0315f;
    const dim3 gNS(D / TILE_N, D / TILE_M);        // 16 x 32 blocks

    for (int h = 0; h < 4; ++h) {
        build_g<<<16384, 256, 0, stream>>>(lA, lB, slotA, sums + h, h);
        finalize_norm<<<1, 1, 0, stream>>>(sums + h, sums + 4 + h, sums + 8 + h);
        scale_x<<<16384, 256, 0, stream>>>(slotA, sums + 8 + h);

        ushort_t* Xc = slotA;
        ushort_t* Xo = slotB;
        for (int s = 0; s < 5; ++s) {
            // A = X @ X^T  (both operands read row-contiguous via TRANSB path)
            gemm_bf16<true,  false><<<gNS, 256, 0, stream>>>(
                Xc, Xc, nullptr, nullptr, Am, D, D, D, 1.0f, 0.0f);
            // B = c*(A @ A^T) + b*A   (A symmetric => A@A^T == A@A)
            gemm_bf16<true,  false><<<gNS, 256, 0, stream>>>(
                Am, Am, Am, nullptr, Bm, D, D, D, nsC, nsB);
            // X' = B @ X + a*X
            gemm_bf16<false, false><<<gNS, 256, 0, stream>>>(
                Bm, Xc, Xc, nullptr, Xo, D, D, D, 1.0f, nsA);
            ushort_t* t = Xc; Xc = Xo; Xo = t;
        }
        add_weff<<<16384, 256, 0, stream>>>(Weff, Xc, sums + 4 + h);
    }

    // out = x @ (W + sum_h alpha_h M_h)^T + bias   (single fused output GEMM)
    const dim3 gOut(D / TILE_N, BS / TILE_M);      // 16 x 64 blocks
    gemm_bf16<true, true><<<gOut, 256, 0, stream>>>(
        xb, Weff, nullptr, bias, d_out, BS, D, D, 1.0f, 0.0f);
}
